// AttentionAttributionTransformerEncoderLayer_13580686590606
// MI455X (gfx1250) — matrix operand order repaired and checked by reading, not compiled, not run on hardware
//
#include <hip/hip_runtime.h>
#include <hip/hip_bf16.h>
#include <math.h>

// ---------------- problem constants ----------------
#define DM     1024                 // d_model
#define NH     16                   // heads
#define DH     64                   // d_head
#define DFF    4096
#define BB     8                    // batch
#define SS     1024                 // seq
#define MTOK   (BB * SS)            // 8192 rows of token matrix
#define LN_EPS 1e-5f

typedef _Float16 v8h  __attribute__((ext_vector_type(8)));
typedef _Float16 v16h __attribute__((ext_vector_type(16)));
typedef float    v8f  __attribute__((ext_vector_type(8)));
typedef unsigned int u32;
typedef u32  u32x4 __attribute__((ext_vector_type(4)));
typedef int  i32x4 __attribute__((ext_vector_type(4)));
typedef int  i32x8 __attribute__((ext_vector_type(8)));

__device__ __forceinline__ v8f wmma16x16x32(v16h a, v16h b, v8f c) {
  // (neg_a, A, neg_b, B, c_mod, C, reuse_a, reuse_b)
  return __builtin_amdgcn_wmma_f32_16x16x32_f16(false, a, false, b, (short)0, c,
                                                false, false);
}

// ---------------------------------------------------------------------------
// TDM: issue a 2D tile load (tile_d0 x tile_d1 elements of 2 bytes) from
// global memory (row stride = stride0 elements) into LDS at byte offset
// lds_off.  D# layout per CDNA5 ISA ch.8 (group0: count/lds/global/type=2,
// group1: data_size=1(2B), tensor dims, tile dims, dim0 stride).
// 6-arg builtin form (clang-23 / therock headers): extra int32x8 group.
// ---------------------------------------------------------------------------
__device__ __forceinline__ void tdm_load_2d_f16(u32 lds_off, const void* gaddr,
                                                u32 tile_d0, u32 tile_d1,
                                                u32 tensor_d0, u32 stride0) {
  const unsigned long long ga = (unsigned long long)gaddr;
  u32x4 g0;
  g0[0] = 1u;                                           // count=1 (valid), user
  g0[1] = lds_off;                                      // lds_addr (bytes)
  g0[2] = (u32)ga;                                      // global_addr[31:0]
  g0[3] = (u32)((ga >> 32) & 0x1FFFFFFu) | (2u << 30);  // addr[56:32] | type=2
  const u32 tensor_d1 = 1u << 20;                       // tall tensor, no OOB
  i32x8 g1;
  g1[0] = (int)(1u << 16);                              // data_size=1 -> 2B
  g1[1] = (int)((tensor_d0 & 0xFFFFu) << 16);           // tensor_dim0 lo16
  g1[2] = (int)((tensor_d0 >> 16) | ((tensor_d1 & 0xFFFFu) << 16));
  g1[3] = (int)((tensor_d1 >> 16) | (tile_d0 << 16));   // tile_dim0
  g1[4] = (int)(tile_d1 & 0xFFFFu);                     // tile_dim1, tile_dim2=0
  g1[5] = (int)stride0;                                 // dim0 stride lo32
  g1[6] = 0;                                            // stride hi, dim1stride
  g1[7] = 0;
  i32x4 z4 = {0, 0, 0, 0};
  i32x8 z8 = {0, 0, 0, 0, 0, 0, 0, 0};
  __builtin_amdgcn_tensor_load_to_lds(g0, g1, z4, z4, z8, 0);
}

// ---------------------------------------------------------------------------
// LDS + TDM double-buffered GEMM core (block tile 128x128, 8 waves as 2x4,
// wave tile 64x32 = 4x2 WMMA accum tiles, K step 32).
// A [M x K] f16 row-major, B [N x K] f16 row-major (C = A @ B^T).
// Wave 0 posts the TDM descriptors for the next K-slab while all waves run
// WMMAs on the current slab out of LDS.
// ---------------------------------------------------------------------------
__device__ __forceinline__ void gemm_lds_tdm(const _Float16* __restrict__ A, int lda,
                                             const _Float16* __restrict__ B, int ldb,
                                             int K, int mBlockBase, int nBlockBase,
                                             v8f (&acc)[4][2]) {
  __shared__ _Float16 lA[2 * 128 * 32];
  __shared__ _Float16 lB[2 * 128 * 32];
  const int tid  = threadIdx.x;
  const int wid  = tid >> 5;
  const int lane = tid & 31;
  const int wm   = wid >> 2, wn = wid & 3;
  const int half = lane >> 4, r = lane & 15;
  const u32 ldsA = (u32)(uintptr_t)&lA[0];
  const u32 ldsB = (u32)(uintptr_t)&lB[0];
  const u32 TILE_BYTES = 128 * 32 * 2;   // 8 KB per buffer

  // prologue: fetch K-slab 0 into buffer 0
  if (wid == 0) {
    tdm_load_2d_f16(ldsA, A + (size_t)mBlockBase * lda, 32, 128, (u32)K, (u32)lda);
    tdm_load_2d_f16(ldsB, B + (size_t)nBlockBase * ldb, 32, 128, (u32)K, (u32)ldb);
    __builtin_amdgcn_s_wait_tensorcnt(0);
  }
  __syncthreads();

  int cur = 0;
  for (int k0 = 0; k0 < K; k0 += 32) {
    // stream next K-slab into the other buffer (overlaps with WMMAs below)
    const bool more = (k0 + 32) < K;
    if (more && wid == 0) {
      const u32 off = (u32)(cur ^ 1) * TILE_BYTES;
      tdm_load_2d_f16(ldsA + off, A + (size_t)mBlockBase * lda + k0 + 32, 32, 128,
                      (u32)K, (u32)lda);
      tdm_load_2d_f16(ldsB + off, B + (size_t)nBlockBase * ldb + k0 + 32, 32, 128,
                      (u32)K, (u32)ldb);
    }
    // compute from current buffer
    const _Float16* sA = lA + cur * (128 * 32);
    const _Float16* sB = lB + cur * (128 * 32);
    v16h af[4];
    v16h bf[2];
#pragma unroll
    for (int i = 0; i < 4; ++i) {
      const _Float16* ap = sA + (wm * 64 + i * 16 + r) * 32 + half * 16;
      v8h lo = *(const v8h*)ap;        // K = +0..7   (lanes16-31: +16..23)
      v8h hi = *(const v8h*)(ap + 8);  // K = +8..15  (lanes16-31: +24..31)
#pragma unroll
      for (int t = 0; t < 8; ++t) { af[i][t] = lo[t]; af[i][t + 8] = hi[t]; }
    }
#pragma unroll
    for (int j = 0; j < 2; ++j)
      bf[j] = *(const v16h*)(sB + (wn * 32 + j * 16 + r) * 32 + half * 16);
#pragma unroll
    for (int i = 0; i < 4; ++i)
#pragma unroll
      for (int j = 0; j < 2; ++j)
        acc[i][j] = wmma16x16x32(af[i], bf[j], acc[i][j]);

    if (more) {
      __syncthreads();                                   // all reads of cur done
      if (wid == 0) __builtin_amdgcn_s_wait_tensorcnt(0);// next slab landed
      __syncthreads();                                   // release consumers
      cur ^= 1;
    }
  }
}

// ---------------------------------------------------------------------------
// direct-from-global GEMM core (used for the batched attention GEMMs).
// ---------------------------------------------------------------------------
template <int MT, int NT>
__device__ __forceinline__ void gemm_core_f16(const _Float16* __restrict__ A, int lda,
                                              const _Float16* __restrict__ B, int ldb,
                                              int K, int mBase, int nBase,
                                              v8f (&acc)[MT][NT]) {
  const int lane = threadIdx.x & 31;
  const int half = lane >> 4;
  const int r    = lane & 15;
  for (int k0 = 0; k0 < K; k0 += 32) {
    v16h af[MT];
    v16h bf[NT];
#pragma unroll
    for (int i = 0; i < MT; ++i) {
      const _Float16* ap = A + (size_t)(mBase + i * 16 + r) * lda + k0 + half * 16;
      v8h lo = *(const v8h*)ap;
      v8h hi = *(const v8h*)(ap + 8);
      __builtin_prefetch(ap + 32, 0, 1);
#pragma unroll
      for (int t = 0; t < 8; ++t) { af[i][t] = lo[t]; af[i][t + 8] = hi[t]; }
    }
#pragma unroll
    for (int j = 0; j < NT; ++j) {
      const _Float16* bp = B + (size_t)(nBase + j * 16 + r) * ldb + k0 + half * 16;
      bf[j] = *(const v16h*)bp;
      __builtin_prefetch(bp + 32, 0, 1);
    }
#pragma unroll
    for (int i = 0; i < MT; ++i)
#pragma unroll
      for (int j = 0; j < NT; ++j)
        acc[i][j] = wmma16x16x32(af[i], bf[j], acc[i][j]);
  }
}

// A in f32 (attention probabilities), converted on load.
template <int MT, int NT>
__device__ __forceinline__ void gemm_core_a32(const float* __restrict__ A, int lda,
                                              const _Float16* __restrict__ B, int ldb,
                                              int K, int mBase, int nBase,
                                              v8f (&acc)[MT][NT]) {
  const int lane = threadIdx.x & 31;
  const int half = lane >> 4;
  const int r    = lane & 15;
  for (int k0 = 0; k0 < K; k0 += 32) {
    v16h af[MT];
    v16h bf[NT];
#pragma unroll
    for (int i = 0; i < MT; ++i) {
      const float* ap = A + (size_t)(mBase + i * 16 + r) * lda + k0 + half * 16;
      __builtin_prefetch(ap + 32, 0, 1);
#pragma unroll
      for (int t = 0; t < 8; ++t) {
        af[i][t]     = (_Float16)ap[t];
        af[i][t + 8] = (_Float16)ap[t + 8];
      }
    }
#pragma unroll
    for (int j = 0; j < NT; ++j) {
      const _Float16* bp = B + (size_t)(nBase + j * 16 + r) * ldb + k0 + half * 16;
      bf[j] = *(const v16h*)bp;
    }
#pragma unroll
    for (int i = 0; i < MT; ++i)
#pragma unroll
      for (int j = 0; j < NT; ++j)
        acc[i][j] = wmma16x16x32(af[i], bf[j], acc[i][j]);
  }
}

// ---------------------------------------------------------------------------
// elementwise f32 -> f16
// ---------------------------------------------------------------------------
__global__ void k_f32_to_f16(const float* __restrict__ in, _Float16* __restrict__ out,
                             int n) {
  int i = blockIdx.x * blockDim.x + threadIdx.x;
  if (i < n) out[i] = (_Float16)in[i];
}

// ---------------------------------------------------------------------------
// QKV projection (TDM core): [8192 x 3072] = x_f16 @ in_proj_w^T + b, scatters
// q[b,h,s,d], k[b,h,s,d] (f16) and vT[b,h,d,s] (f16). grid (24,64), 256 thr.
// ---------------------------------------------------------------------------
__global__ void k_qkv(const _Float16* __restrict__ xh, const _Float16* __restrict__ w,
                      const float* __restrict__ bias, _Float16* __restrict__ qh,
                      _Float16* __restrict__ kh, _Float16* __restrict__ vT) {
  const int lane = threadIdx.x & 31;
  const int wid  = threadIdx.x >> 5;
  const int wm   = wid >> 2, wn = wid & 3;
  const int mBase = blockIdx.y * 128 + wm * 64;
  const int nBase = blockIdx.x * 128 + wn * 32;
  v8f acc[4][2] = {};
  gemm_lds_tdm(xh, DM, w, DM, DM, blockIdx.y * 128, blockIdx.x * 128, acc);
  const int half = lane >> 4, r = lane & 15;
#pragma unroll
  for (int i = 0; i < 4; ++i)
#pragma unroll
    for (int j = 0; j < 2; ++j) {
      const int n    = nBase + j * 16 + r;
      const float bv = bias[n];
#pragma unroll
      for (int t = 0; t < 8; ++t) {
        const int   m   = mBase + i * 16 + t + 8 * half;
        const float val = acc[i][j][t] + bv;
        const int   b = m >> 10, s = m & (SS - 1);
        if (n < DM) {
          const int h = n >> 6, d = n & 63;
          qh[(((size_t)(b * NH + h)) * SS + s) * DH + d] = (_Float16)val;
        } else if (n < 2 * DM) {
          const int n2 = n - DM, h = n2 >> 6, d = n2 & 63;
          kh[(((size_t)(b * NH + h)) * SS + s) * DH + d] = (_Float16)val;
        } else {
          const int n2 = n - 2 * DM, h = n2 >> 6, d = n2 & 63;
          vT[(((size_t)(b * NH + h)) * DH + d) * SS + s] = (_Float16)val;
        }
      }
    }
}

// ---------------------------------------------------------------------------
// scores = q @ k^T * scale, batched over z = b*NH+h. grid (8, 8, 128), 256 thr.
// ---------------------------------------------------------------------------
__global__ void k_scores(const _Float16* __restrict__ q, const _Float16* __restrict__ k,
                         float* __restrict__ attn, float scale) {
  const size_t z = blockIdx.z;
  const _Float16* A = q + z * (size_t)SS * DH;
  const _Float16* B = k + z * (size_t)SS * DH;
  float* C = attn + z * (size_t)SS * SS;
  const int lane = threadIdx.x & 31;
  const int wid  = threadIdx.x >> 5;
  const int wm = wid >> 2, wn = wid & 3;
  const int mBase = blockIdx.y * 128 + wm * 64;
  const int nBase = blockIdx.x * 128 + wn * 32;
  v8f acc[4][2] = {};
  gemm_core_f16<4, 2>(A, DH, B, DH, DH, mBase, nBase, acc);
  const int half = lane >> 4, r = lane & 15;
#pragma unroll
  for (int i = 0; i < 4; ++i)
#pragma unroll
    for (int j = 0; j < 2; ++j) {
      const int n = nBase + j * 16 + r;
#pragma unroll
      for (int t = 0; t < 8; ++t) {
        const int m = mBase + i * 16 + t + 8 * half;
        C[(size_t)m * SS + n] = acc[i][j][t] * scale;
      }
    }
}

// ---------------------------------------------------------------------------
// row softmax in place, * head_mask. one block (256 thr) per row of 1024.
// ---------------------------------------------------------------------------
__global__ void k_softmax(float* __restrict__ attn, const float* __restrict__ hm) {
  __shared__ float red[256];
  const size_t row = (size_t)blockIdx.y * SS + blockIdx.x;
  float* p = attn + row * SS;
  const int tid = threadIdx.x;
  float v[4];
  float mx = -3.4e38f;
#pragma unroll
  for (int t = 0; t < 4; ++t) { v[t] = p[tid + t * 256]; mx = fmaxf(mx, v[t]); }
  red[tid] = mx;
  __syncthreads();
  for (int s = 128; s > 0; s >>= 1) {
    if (tid < s) red[tid] = fmaxf(red[tid], red[tid + s]);
    __syncthreads();
  }
  mx = red[0];
  __syncthreads();
  float sum = 0.f;
#pragma unroll
  for (int t = 0; t < 4; ++t) { v[t] = __expf(v[t] - mx); sum += v[t]; }
  red[tid] = sum;
  __syncthreads();
  for (int s = 128; s > 0; s >>= 1) {
    if (tid < s) red[tid] += red[tid + s];
    __syncthreads();
  }
  const float inv = hm[0] / red[0];
#pragma unroll
  for (int t = 0; t < 4; ++t) p[tid + t * 256] = v[t] * inv;
}

// ---------------------------------------------------------------------------
// attn @ v : [1024 x 64] per (b,h); A f32 converted on the fly, B = vT (f16).
// scatters into ctx[b,s, h*64+d] f16. grid (8, 128), 128 thr (4 waves 2x2).
// ---------------------------------------------------------------------------
__global__ void k_av(const float* __restrict__ attn, const _Float16* __restrict__ vT,
                     _Float16* __restrict__ ctx) {
  const size_t z = blockIdx.y;
  const int   b = (int)(z >> 4), h = (int)(z & 15);
  const float*    A = attn + z * (size_t)SS * SS;
  const _Float16* B = vT + z * (size_t)DH * SS;
  const int lane = threadIdx.x & 31;
  const int wid  = threadIdx.x >> 5;
  const int wm = wid >> 1, wn = wid & 1;
  const int mBase = blockIdx.x * 128 + wm * 64;
  const int nBase = wn * 32;
  v8f acc[4][2] = {};
  gemm_core_a32<4, 2>(A, SS, B, SS, SS, mBase, nBase, acc);
  const int half = lane >> 4, r = lane & 15;
#pragma unroll
  for (int i = 0; i < 4; ++i)
#pragma unroll
    for (int j = 0; j < 2; ++j) {
      const int d = nBase + j * 16 + r;
#pragma unroll
      for (int t = 0; t < 8; ++t) {
        const int s = mBase + i * 16 + t + 8 * half;
        ctx[((size_t)(b * SS + s)) * DM + h * DH + d] = (_Float16)acc[i][j][t];
      }
    }
}

// ---------------------------------------------------------------------------
// generic GEMM (TDM core) + bias -> f32. grid (N/128, M/128), 256 thr.
// ---------------------------------------------------------------------------
__global__ void k_gemm_bias(const _Float16* __restrict__ A, int lda,
                            const _Float16* __restrict__ B, int ldb,
                            const float* __restrict__ bias, float* __restrict__ C,
                            int ldc, int K) {
  const int lane = threadIdx.x & 31;
  const int wid  = threadIdx.x >> 5;
  const int wm = wid >> 2, wn = wid & 3;
  const int mBase = blockIdx.y * 128 + wm * 64;
  const int nBase = blockIdx.x * 128 + wn * 32;
  v8f acc[4][2] = {};
  gemm_lds_tdm(A, lda, B, ldb, K, blockIdx.y * 128, blockIdx.x * 128, acc);
  const int half = lane >> 4, r = lane & 15;
#pragma unroll
  for (int i = 0; i < 4; ++i)
#pragma unroll
    for (int j = 0; j < 2; ++j) {
      const int   n  = nBase + j * 16 + r;
      const float bv = bias[n];
#pragma unroll
      for (int t = 0; t < 8; ++t) {
        const int m = mBase + i * 16 + t + 8 * half;
        C[(size_t)m * ldc + n] = acc[i][j][t] + bv;
      }
    }
}

// ---------------------------------------------------------------------------
// GEMM (TDM core) + bias + ReLU -> f16 (FFN layer 1). grid (N/128, M/128).
// ---------------------------------------------------------------------------
__global__ void k_gemm_bias_relu_f16(const _Float16* __restrict__ A, int lda,
                                     const _Float16* __restrict__ B, int ldb,
                                     const float* __restrict__ bias,
                                     _Float16* __restrict__ C, int ldc, int K) {
  const int lane = threadIdx.x & 31;
  const int wid  = threadIdx.x >> 5;
  const int wm = wid >> 2, wn = wid & 3;
  const int mBase = blockIdx.y * 128 + wm * 64;
  const int nBase = blockIdx.x * 128 + wn * 32;
  v8f acc[4][2] = {};
  gemm_lds_tdm(A, lda, B, ldb, K, blockIdx.y * 128, blockIdx.x * 128, acc);
  const int half = lane >> 4, r = lane & 15;
#pragma unroll
  for (int i = 0; i < 4; ++i)
#pragma unroll
    for (int j = 0; j < 2; ++j) {
      const int   n  = nBase + j * 16 + r;
      const float bv = bias[n];
#pragma unroll
      for (int t = 0; t < 8; ++t) {
        const int   m = mBase + i * 16 + t + 8 * half;
        const float y = fmaxf(acc[i][j][t] + bv, 0.f);
        C[(size_t)m * ldc + n] = (_Float16)y;
      }
    }
}

// ---------------------------------------------------------------------------
// out = LayerNorm(a + b) * g + be ; optional f16 copy. one block per row of 1024.
// ---------------------------------------------------------------------------
__global__ void k_add_ln(const float* __restrict__ a, const float* __restrict__ b,
                         const float* __restrict__ g, const float* __restrict__ be,
                         float* __restrict__ out, _Float16* __restrict__ out16) {
  __shared__ float red[256];
  const size_t row = blockIdx.x;
  const float* pa = a + row * DM;
  const float* pb = b + row * DM;
  const int tid = threadIdx.x;
  float v[4];
  float s1 = 0.f, s2 = 0.f;
#pragma unroll
  for (int t = 0; t < 4; ++t) {
    v[t] = pa[tid + t * 256] + pb[tid + t * 256];
    s1 += v[t];
    s2 += v[t] * v[t];
  }
  red[tid] = s1;
  __syncthreads();
  for (int s = 128; s > 0; s >>= 1) {
    if (tid < s) red[tid] += red[tid + s];
    __syncthreads();
  }
  const float mean = red[0] * (1.f / DM);
  __syncthreads();
  red[tid] = s2;
  __syncthreads();
  for (int s = 128; s > 0; s >>= 1) {
    if (tid < s) red[tid] += red[tid + s];
    __syncthreads();
  }
  const float var = red[0] * (1.f / DM) - mean * mean;
  const float inv = rsqrtf(var + LN_EPS);
#pragma unroll
  for (int t = 0; t < 4; ++t) {
    const int   c = tid + t * 256;
    const float y = (v[t] - mean) * inv * g[c] + be[c];
    out[row * DM + c] = y;
    if (out16) out16[row * DM + c] = (_Float16)y;
  }
}

// ---------------------------------------------------------------------------
// host side
// ---------------------------------------------------------------------------
extern "C" void kernel_launch(void* const* d_in, const int* in_sizes, int n_in,
                              void* d_out, int out_size, void* d_ws, size_t ws_size,
                              hipStream_t stream) {
  (void)in_sizes; (void)n_in; (void)out_size; (void)ws_size;
  const float* x     = (const float*)d_in[0];
  const float* inw   = (const float*)d_in[1];
  const float* inb   = (const float*)d_in[2];
  const float* outw  = (const float*)d_in[3];
  const float* outb  = (const float*)d_in[4];
  const float* ln1w  = (const float*)d_in[5];
  const float* ln1b  = (const float*)d_in[6];
  const float* ln2w  = (const float*)d_in[7];
  const float* ln2b  = (const float*)d_in[8];
  const float* w1    = (const float*)d_in[9];
  const float* b1    = (const float*)d_in[10];
  const float* w2    = (const float*)d_in[11];
  const float* b2    = (const float*)d_in[12];
  const float* hmask = (const float*)d_in[13];

  float* out  = (float*)d_out;
  float* attn = out + (size_t)MTOK * DM;   // output tuple: (out, attn)

  // --- carve workspace (256B-aligned slices) ---
  char* w = (char*)d_ws;
  auto alloc = [&](size_t bytes) -> void* {
    void* p = (void*)w;
    w += (bytes + 255) & ~(size_t)255;
    return p;
  };
  _Float16* xh    = (_Float16*)alloc((size_t)MTOK * DM * 2);
  _Float16* wqkvh = (_Float16*)alloc((size_t)3 * DM * DM * 2);
  _Float16* wouth = (_Float16*)alloc((size_t)DM * DM * 2);
  _Float16* w1h   = (_Float16*)alloc((size_t)DFF * DM * 2);
  _Float16* w2h   = (_Float16*)alloc((size_t)DM * DFF * 2);
  _Float16* qh    = (_Float16*)alloc((size_t)BB * NH * SS * DH * 2);
  _Float16* kh    = (_Float16*)alloc((size_t)BB * NH * SS * DH * 2);
  _Float16* vTh   = (_Float16*)alloc((size_t)BB * NH * DH * SS * 2);
  _Float16* ctxh  = (_Float16*)alloc((size_t)MTOK * DM * 2);
  float*    proj  = (float*)alloc((size_t)MTOK * DM * 4);
  float*    x1    = (float*)alloc((size_t)MTOK * DM * 4);
  _Float16* x1h   = (_Float16*)alloc((size_t)MTOK * DM * 2);
  _Float16* hbuf  = (_Float16*)alloc((size_t)MTOK * DFF * 2);
  float*    ff    = (float*)alloc((size_t)MTOK * DM * 4);

  auto cvt = [&](const float* src, _Float16* dst, int n) {
    k_f32_to_f16<<<(n + 255) / 256, 256, 0, stream>>>(src, dst, n);
  };
  cvt(x, xh, MTOK * DM);
  cvt(inw, wqkvh, 3 * DM * DM);
  cvt(outw, wouth, DM * DM);
  cvt(w1, w1h, DFF * DM);
  cvt(w2, w2h, DM * DFF);

  // QKV projection + scatter
  k_qkv<<<dim3(3 * DM / 128, MTOK / 128), 256, 0, stream>>>(xh, wqkvh, inb, qh, kh, vTh);

  // scores = q k^T / sqrt(64)
  k_scores<<<dim3(SS / 128, SS / 128, BB * NH), 256, 0, stream>>>(qh, kh, attn, 0.125f);

  // softmax rows, * head_mask (in place in d_out attn region)
  k_softmax<<<dim3(SS, BB * NH), 256, 0, stream>>>(attn, hmask);

  // attn @ v -> ctx (f16, [B,S,D] layout)
  k_av<<<dim3(SS / 128, BB * NH), 128, 0, stream>>>(attn, vTh, ctxh);

  // out projection
  k_gemm_bias<<<dim3(DM / 128, MTOK / 128), 256, 0, stream>>>(ctxh, DM, wouth, DM,
                                                              outb, proj, DM, DM);
  // x1 = LN1(x + proj)  (+ f16 copy)
  k_add_ln<<<MTOK, 256, 0, stream>>>(x, proj, ln1w, ln1b, x1, x1h);

  // h = relu(x1 W1^T + b1)  (f16)
  k_gemm_bias_relu_f16<<<dim3(DFF / 128, MTOK / 128), 256, 0, stream>>>(
      x1h, DM, w1h, DM, b1, hbuf, DFF, DM);

  // ff = h W2^T + b2
  k_gemm_bias<<<dim3(DM / 128, MTOK / 128), 256, 0, stream>>>(hbuf, DFF, w2h, DFF,
                                                              b2, ff, DM, DFF);
  // out = LN2(x1 + ff)
  k_add_ln<<<MTOK, 256, 0, stream>>>(x1, ff, ln2w, ln2b, out, (_Float16*)nullptr);
}
